// PathNetWithMP_61838939128053
// MI455X (gfx1250) — matrix operand order repaired and checked by reading, not compiled, not run on hardware
//
#include <hip/hip_runtime.h>

// ---------------------------------------------------------------------------
// PathNetWithMP forward for MI455X (gfx1250, wave32, WMMA).
// Dense linear algebra: bf16 WMMA GEMM (v_wmma_f32_16x16x32_bf16, f32 accum).
// Weights + activations are pre-converted to bf16 so the GEMM inner loop is
// pure load/ds/wmma. Conv1d = 3 shifted GEMM passes sharing one converted A.
// GEMM requires K % 32 == 0 (all call sites: K in {256, 512}).
// ---------------------------------------------------------------------------

#define H 256
#define GRAPHS 64
#define BM 64
#define BN 64
#define BK 32

typedef __attribute__((ext_vector_type(16))) __bf16 v16bf;
typedef __attribute__((ext_vector_type(8)))  float  v8f;
typedef unsigned short ushort_t;

union FragAB { v16bf v; unsigned u[8]; };

// RNE f32->bf16 pair pack: round both, then v_perm the two high halves.
__device__ __forceinline__ unsigned pack2bf(float lo, float hi) {
  unsigned ul = __float_as_uint(lo);
  unsigned uh = __float_as_uint(hi);
  ul += 0x7FFFu + ((ul >> 16) & 1u);
  uh += 0x7FFFu + ((uh >> 16) & 1u);
  return __builtin_amdgcn_perm(uh, ul, 0x07060302u);  // {uh[3],uh[2],ul[3],ul[2]}
}

// ---------------------------------------------------------------------------
// f32 -> bf16 converter (n must be a multiple of 4; true for all buffers)
// ---------------------------------------------------------------------------
__global__ void f32_to_bf16_kernel(const float* __restrict__ in,
                                   ushort_t* __restrict__ out, long long n) {
  long long i  = ((long long)blockIdx.x * blockDim.x + threadIdx.x) * 4;
  long long st = (long long)gridDim.x * blockDim.x * 4;
  for (; i < n; i += st) {
    float4 v = *(const float4*)(in + i);
    unsigned* o = (unsigned*)(out + i);
    o[0] = pack2bf(v.x, v.y);
    o[1] = pack2bf(v.z, v.w);
  }
}

// ---------------------------------------------------------------------------
// WMMA GEMM: Cout = epilogue(A * B + ...)
//   A: [M,K] bf16 row-major (optionally row-shifted within periods of kPeriod
//      for the path conv; out-of-period rows contribute zero = SAME padding)
//   B: [K,N] bf16 row-major
// epilogue modes:
//   0: Cout = acc + bias            (bias may be null -> 0)
//   1: Cout = relu(acc + bias)
//   2: Cout = addsrc + relu(acc + bias)
//   3: Cout = Cin + acc                         (partial accumulate)
//   4: Cout = relu(Cin + acc + bias)
//   5: Cout = relu(addsrc + Cin + acc + bias)
// BVEC: compile-time flag — vectorized B loads (requires N % 8 == 0).
// ---------------------------------------------------------------------------
template <bool BVEC>
__global__ __launch_bounds__(256) void gemm_wmma_kernel(
    const ushort_t* __restrict__ A, const ushort_t* __restrict__ Bm,
    const float* __restrict__ bias, const float* __restrict__ addsrc,
    const float* __restrict__ Cin, float* __restrict__ Cout,
    int M, int N, int K, int kPeriod, int shift, int mode)
{
  __shared__ ushort_t As[BM][BK + 2];   // row stride 68B (17 dwords)
  __shared__ ushort_t Bs[BN][BK + 2];   // stored [n][k] (K contiguous)

  const int tid  = threadIdx.x;
  const int bm0  = blockIdx.x * BM;
  const int bn0  = blockIdx.y * BN;

  const int wave = tid >> 5;
  const int lane = tid & 31;
  const int wy   = wave >> 1;      // 0..3 : 16-row strip
  const int wx   = wave & 1;       // 0..1 : 32-col strip
  const int half = lane >> 4;      // 0/1
  const int m16  = lane & 15;

  v8f accs[2];
  accs[0] = (v8f){0,0,0,0,0,0,0,0};
  accs[1] = (v8f){0,0,0,0,0,0,0,0};

  // A tile load mapping: 4 threads per row, 8 consecutive bf16 each (1x b128)
  const int lrA = tid >> 2;             // 0..63
  const int lkA = (tid & 3) << 3;       // 0,8,16,24
  int  growA  = bm0 + lrA;
  int  arow   = growA;
  bool avalid = (growA < M);
  if (kPeriod > 0 && avalid) {
    int pos = growA % kPeriod;
    int np  = pos + shift;
    arow    = growA + shift;
    avalid  = (np >= 0) && (np < kPeriod) && (arow >= 0) && (arow < M);
  }

  // B tile load mapping: 8 threads per k-row, 8 consecutive bf16 each (1x b128)
  const int lkB = tid >> 3;             // 0..31
  const int lnB = (tid & 7) << 3;       // 0,8,..,56
  const int gnB = bn0 + lnB;
  const bool bvalid = (gnB + 8 <= N);

  // Explicitly advanced pointers: per-iteration strides are constant, so the
  // prefetch targets fold into immediate offsets (no per-iter 64-bit mul).
  const ushort_t* aPtr = A + (long long)arow * K + lkA;
  const ushort_t* bPtr = Bm + (long long)lkB * N + gnB;
  const long long bStep = (long long)BK * N;

  for (int k0 = 0; k0 < K; k0 += BK) {
    // ---- A tile (raw bf16 copy through LDS) ----
    {
      uint4 av = {0, 0, 0, 0};
      if (avalid) {
        av = *(const uint4*)aPtr;
        __builtin_prefetch(aPtr + BK, 0, 1);       // global_prefetch_b8
      }
      unsigned* dA = (unsigned*)&As[lrA][lkA];
      dA[0] = av.x; dA[1] = av.y; dA[2] = av.z; dA[3] = av.w;
    }
    // ---- B tile (store transposed [n][k]) ----
    if (BVEC) {
      uint4 bv = {0, 0, 0, 0};
      if (bvalid) {
        bv = *(const uint4*)bPtr;
        __builtin_prefetch(bPtr + bStep, 0, 1);    // global_prefetch_b8
      }
      Bs[lnB + 0][lkB] = (ushort_t)(bv.x);
      Bs[lnB + 1][lkB] = (ushort_t)(bv.x >> 16);
      Bs[lnB + 2][lkB] = (ushort_t)(bv.y);
      Bs[lnB + 3][lkB] = (ushort_t)(bv.y >> 16);
      Bs[lnB + 4][lkB] = (ushort_t)(bv.z);
      Bs[lnB + 5][lkB] = (ushort_t)(bv.z >> 16);
      Bs[lnB + 6][lkB] = (ushort_t)(bv.w);
      Bs[lnB + 7][lkB] = (ushort_t)(bv.w >> 16);
    } else {
#pragma unroll
      for (int e = 0; e < 8; ++e) {
        ushort_t v = 0;
        if (gnB + e < N) v = bPtr[e];
        Bs[lnB + e][lkB] = v;
      }
    }
    aPtr += BK;
    bPtr += bStep;
    __syncthreads();

    FragAB fa, fb0, fb1;
    const int arL = wy * 16 + m16;
#pragma unroll
    for (int p = 0; p < 8; ++p) {
      // 16-bit A 16x32 layout: elem j: K = (j/8)*16 + half*8 + (j%8)
      int kb = ((p >> 2) << 4) + (half << 3) + ((p << 1) & 7);
      fa.u[p] = *(const unsigned*)&As[arL][kb];
    }
    const int b0r = wx * 32 + m16;
    const int b1r = wx * 32 + 16 + m16;
#pragma unroll
    for (int p = 0; p < 8; ++p) {
      // 16-bit B 32x16 layout: elem j: K = (j/8)*16 + half*8 + (j%8)
      int kb = ((p >> 2) << 4) + (half << 3) + ((p << 1) & 7);
      fb0.u[p] = *(const unsigned*)&Bs[b0r][kb];
      fb1.u[p] = *(const unsigned*)&Bs[b1r][kb];
    }

    accs[0] = __builtin_amdgcn_wmma_f32_16x16x32_bf16(
        false, fa.v, false, fb0.v, (short)0, accs[0], false, false);
    accs[1] = __builtin_amdgcn_wmma_f32_16x16x32_bf16(
        false, fa.v, false, fb1.v, (short)0, accs[1], false, false);

    __syncthreads();
  }

#pragma unroll
  for (int cb = 0; cb < 2; ++cb) {
    int col = bn0 + wx * 32 + cb * 16 + m16;
    if (col >= N) continue;
    float bv = bias ? bias[col] : 0.0f;
#pragma unroll
    for (int r = 0; r < 8; ++r) {
      int row = bm0 + wy * 16 + (half << 3) + r;   // C layout: VGPR r -> M=r+8*half
      if (row >= M) continue;
      long long idx = (long long)row * N + col;
      float v = accs[cb][r];
      float o;
      switch (mode) {
        case 0:  o = v + bv; break;
        case 1:  o = fmaxf(v + bv, 0.0f); break;
        case 2:  o = addsrc[idx] + fmaxf(v + bv, 0.0f); break;
        case 3:  o = Cin[idx] + v; break;
        case 4:  o = fmaxf(Cin[idx] + v + bv, 0.0f); break;
        default: o = fmaxf(addsrc[idx] + Cin[idx] + v + bv, 0.0f); break;
      }
      Cout[idx] = o;
    }
  }
}

// --------------------------- support kernels -------------------------------

__global__ void zero_kernel(float* p, long long n) {
  long long i  = (long long)blockIdx.x * blockDim.x + threadIdx.x;
  long long st = (long long)gridDim.x * blockDim.x;
  for (; i < n; i += st) p[i] = 0.0f;
}

__global__ void gather_rows_kernel(const float* __restrict__ tab,
                                   const int* __restrict__ idx,
                                   float* __restrict__ out) {
  int r = blockIdx.x, h = threadIdx.x;
  out[(long long)r * H + h] = tab[(long long)idx[r] * H + h];
}

__global__ void gine_init_kernel(const float* __restrict__ x,
                                 const float* __restrict__ eps,
                                 float* __restrict__ z, long long n) {
  float s = 1.0f + eps[0];
  long long i  = (long long)blockIdx.x * blockDim.x + threadIdx.x;
  long long st = (long long)gridDim.x * blockDim.x;
  for (; i < n; i += st) z[i] = s * x[i];
}

__global__ void gine_edge_kernel(const float* __restrict__ x,
                                 const float* __restrict__ bond,
                                 const int* __restrict__ src,
                                 const int* __restrict__ dst,
                                 const int* __restrict__ attr,
                                 float* __restrict__ z) {
  int e = blockIdx.x, h = threadIdx.x;
  int s0 = src[e], d0 = dst[e], a0 = attr[e];
  float v = x[(long long)s0 * H + h] + bond[(long long)a0 * H + h];
  atomicAdd(&z[(long long)d0 * H + h], fmaxf(v, 0.0f));
}

// sums[sidx[m]] += vsrc[gidx[m]], cnts[sidx[m]] += 1
__global__ void scatter_mean_kernel(const float* __restrict__ vsrc,
                                    const int* __restrict__ gidx,
                                    const int* __restrict__ sidx,
                                    float* __restrict__ sums,
                                    float* __restrict__ cnts) {
  int m = blockIdx.x, h = threadIdx.x;
  int g = gidx[m], s0 = sidx[m];
  atomicAdd(&sums[(long long)s0 * H + h], vsrc[(long long)g * H + h]);
  if (h == 0) atomicAdd(&cnts[s0], 1.0f);
}

__global__ void seg_rows_kernel(const float* __restrict__ x,
                                const int* __restrict__ seg,
                                float* __restrict__ sums,
                                float* __restrict__ cnts) {
  int i = blockIdx.x, h = threadIdx.x;
  int g = seg[i];
  atomicAdd(&sums[(long long)g * H + h], x[(long long)i * H + h]);
  if (h == 0) atomicAdd(&cnts[g], 1.0f);
}

__global__ void seg_div_kernel(float* __restrict__ sums,
                               const float* __restrict__ cnts) {
  int s0 = blockIdx.x, h = threadIdx.x;
  sums[(long long)s0 * H + h] /= fmaxf(cnts[s0], 1.0f);
}

// per-column mean / rsqrt(var+eps) over M rows (block per column)
__global__ void bn_stats_kernel(const float* __restrict__ x, int M, int C,
                                float* __restrict__ mean,
                                float* __restrict__ rstd) {
  int c = blockIdx.x, t = threadIdx.x;
  float s = 0.0f, s2 = 0.0f;
  for (int r = t; r < M; r += 256) {
    float v = x[(long long)r * C + c];
    s += v; s2 += v * v;
  }
  __shared__ float ss[256], qq[256];
  ss[t] = s; qq[t] = s2;
  __syncthreads();
  for (int o = 128; o > 0; o >>= 1) {
    if (t < o) { ss[t] += ss[t + o]; qq[t] += qq[t + o]; }
    __syncthreads();
  }
  if (t == 0) {
    float m  = ss[0] / (float)M;
    float v  = qq[0] / (float)M - m * m;
    mean[c] = m;
    rstd[c] = rsqrtf(fmaxf(v, 0.0f) + 1e-5f);
  }
}

// y = relu((x-mean)*rstd*g + b)   (block per row, C threads)
__global__ void bn_apply_kernel(const float* __restrict__ x,
                                const float* __restrict__ mean,
                                const float* __restrict__ rstd,
                                const float* __restrict__ g,
                                const float* __restrict__ b,
                                float* __restrict__ y, int C) {
  int r = blockIdx.x, c = threadIdx.x;
  long long idx = (long long)r * C + c;
  float v = (x[idx] - mean[c]) * rstd[c] * g[c] + b[c];
  y[idx] = fmaxf(v, 0.0f);
}

// --------------------------- host orchestration ----------------------------

static inline void launch_gemm(const ushort_t* A, const ushort_t* B,
                               const float* bias, const float* addsrc,
                               const float* Cin, float* Cout,
                               int M, int N, int K, int kPeriod, int shift,
                               int mode, hipStream_t s) {
  dim3 g((M + BM - 1) / BM, (N + BN - 1) / BN);
  if ((N & 7) == 0) {
    gemm_wmma_kernel<true><<<g, 256, 0, s>>>(A, B, bias, addsrc, Cin, Cout,
                                             M, N, K, kPeriod, shift, mode);
  } else {
    gemm_wmma_kernel<false><<<g, 256, 0, s>>>(A, B, bias, addsrc, Cin, Cout,
                                              M, N, K, kPeriod, shift, mode);
  }
}

static inline void launch_zero(float* p, long long n, hipStream_t s) {
  zero_kernel<<<2048, 256, 0, s>>>(p, n);
}

static inline void launch_cvt(const float* src, ushort_t* dst, long long n,
                              hipStream_t s) {
  f32_to_bf16_kernel<<<1024, 256, 0, s>>>(src, dst, n);
}

extern "C" void kernel_launch(void* const* d_in, const int* in_sizes, int n_in,
                              void* d_out, int out_size, void* d_ws, size_t ws_size,
                              hipStream_t stream) {
  (void)n_in; (void)ws_size;
  auto F = [&](int i) { return (const float*)d_in[i]; };
  auto I = [&](int i) { return (const int*)d_in[i]; };

  // ---- params (jax pytree order: dict keys sorted, lists in order) ----
  int ix = 0;
  const float* atom_emb   = F(ix++);
  const float* atom_lin_b = F(ix++);
  const float* atom_lin_w = F(ix++);

  struct PathP { const float *a2p_b,*a2p_w,*c1_b,*c1_w,*c2_b,*c2_w,*p2a_b,*p2a_w; };
  struct LayerP {
    const float *bn_b,*bn_g,*bond,*eps,*gbn_b,*gbn_g,*l1_b,*l1_w,*l2_b,*l2_w;
    PathP pk[2];
  };
  LayerP Lp[3];
  for (int l = 0; l < 3; ++l) {
    Lp[l].bn_b  = F(ix++); Lp[l].bn_g  = F(ix++);
    Lp[l].bond  = F(ix++); Lp[l].eps   = F(ix++);
    Lp[l].gbn_b = F(ix++); Lp[l].gbn_g = F(ix++);
    Lp[l].l1_b  = F(ix++); Lp[l].l1_w  = F(ix++);
    Lp[l].l2_b  = F(ix++); Lp[l].l2_w  = F(ix++);
    for (int t = 0; t < 2; ++t) {                 // "3" then "4"
      PathP& p = Lp[l].pk[t];
      p.a2p_b = F(ix++); p.a2p_w = F(ix++);       // a2p: b, w
      p.c1_b  = F(ix++); p.c1_w  = F(ix++);       // conv1: b, w[3,H,H]
      p.c2_b  = F(ix++); p.c2_w  = F(ix++);       // conv2: b, w
      p.p2a_b = F(ix++); p.p2a_w = F(ix++);       // p2a: b, w
    }
  }
  const float* lin_b   = F(ix++);
  const float* lin_w   = F(ix++);
  const float* pemb[2] = { F(ix++), F(ix++) };

  // ---- data inputs ----
  const int ix_atoms = ix;  const int* x_atoms    = I(ix++);
  const int* edge_index = I(ix++);
  const int ix_attr  = ix;  const int* edge_attr  = I(ix++);
  const int* batch = I(ix++);
  const int* xpath[2]; const int* arow[2]; const int* acol[2];
  int slots[2], nmap[2];
  for (int t = 0; t < 2; ++t) {
    slots[t] = in_sizes[ix]; xpath[t] = I(ix++);
    nmap[t]  = in_sizes[ix]; arow[t]  = I(ix++);
    acol[t]  = I(ix++);
  }

  const int N = in_sizes[ix_atoms];
  const int E = in_sizes[ix_attr];
  const int OUT = out_size / GRAPHS;
  const int* e_src = edge_index;
  const int* e_dst = edge_index + E;

  // ---- workspace layout ----
  float* W = (float*)d_ws;
  size_t off = 0;
  auto alloc = [&](size_t n) {
    float* p = W + off; off += n; off = (off + 3) & ~(size_t)3; return p;
  };
  auto allocBF = [&](size_t n) {
    ushort_t* p = (ushort_t*)(W + off);
    off += (n + 1) / 2; off = (off + 3) & ~(size_t)3; return p;
  };
  const size_t NH  = (size_t)N * H;
  const size_t N2H = (size_t)N * 2 * H;
  const size_t HH  = (size_t)H * H;
  size_t SMAX = (size_t)((slots[0] > slots[1]) ? slots[0] : slots[1]) * H;
  if (NH > SMAX) SMAX = NH;
  if (N2H > SMAX) SMAX = N2H;

  float* bx    = alloc(NH);                         // atom features x
  float* bz    = alloc(NH);                         // GINE z / lin2 out
  float* bh    = alloc(N2H);                        // GINE hidden
  float* bxp[2] = { alloc((size_t)slots[0] * H),
                    alloc((size_t)slots[1] * H) };  // path features
  float* bsc   = alloc(SMAX);                       // seg-mean / conv1 hidden
  float* btmp  = alloc(SMAX);                       // conv2 partial
  int cntMax = N;
  if (slots[0] > cntMax) cntMax = slots[0];
  if (slots[1] > cntMax) cntMax = slots[1];
  float* bcnt  = alloc((size_t)cntMax);
  float* bmean = alloc(512);
  float* brstd = alloc(512);
  float* bgs   = alloc((size_t)GRAPHS * H);
  float* bgc   = alloc(GRAPHS);
  float* bxg   = alloc((size_t)GRAPHS * H);

  // bf16 regions: activations (shared) + all weights
  ushort_t* abf = allocBF(SMAX);
  struct PathW { ushort_t *a2p, *c1, *c2, *p2a; };
  struct LayerW { ushort_t *l1, *l2; PathW pk[2]; };
  LayerW Lw[3];
  for (int l = 0; l < 3; ++l) {
    Lw[l].l1 = allocBF(HH * 2);
    Lw[l].l2 = allocBF(HH * 2);
    for (int t = 0; t < 2; ++t) {
      Lw[l].pk[t].a2p = allocBF(HH);
      Lw[l].pk[t].c1  = allocBF(HH * 3);
      Lw[l].pk[t].c2  = allocBF(HH * 3);
      Lw[l].pk[t].p2a = allocBF(HH);
    }
  }
  ushort_t* w_atomlin = allocBF(HH);
  ushort_t* w_lin     = allocBF((size_t)H * OUT);

  // ---- convert all weights to bf16 (cheap, deterministic per call) ----
  for (int l = 0; l < 3; ++l) {
    launch_cvt(Lp[l].l1_w, Lw[l].l1, (long long)HH * 2, stream);
    launch_cvt(Lp[l].l2_w, Lw[l].l2, (long long)HH * 2, stream);
    for (int t = 0; t < 2; ++t) {
      launch_cvt(Lp[l].pk[t].a2p_w, Lw[l].pk[t].a2p, (long long)HH, stream);
      launch_cvt(Lp[l].pk[t].c1_w,  Lw[l].pk[t].c1,  (long long)HH * 3, stream);
      launch_cvt(Lp[l].pk[t].c2_w,  Lw[l].pk[t].c2,  (long long)HH * 3, stream);
      launch_cvt(Lp[l].pk[t].p2a_w, Lw[l].pk[t].p2a, (long long)HH, stream);
    }
  }
  launch_cvt(atom_lin_w, w_atomlin, (long long)HH, stream);
  launch_cvt(lin_w, w_lin, (long long)H * OUT, stream);

  // ---- embeddings ----
  gather_rows_kernel<<<N, H, 0, stream>>>(atom_emb, x_atoms, bx);
  gather_rows_kernel<<<slots[0], H, 0, stream>>>(pemb[0], xpath[0], bxp[0]);
  gather_rows_kernel<<<slots[1], H, 0, stream>>>(pemb[1], xpath[1], bxp[1]);

  for (int l = 0; l < 3; ++l) {
    const LayerP& L = Lp[l];
    const LayerW& Wt = Lw[l];

    // ---- GINEConv: z = (1+eps)*x + segsum(relu(x[src]+bond[attr]), dst) ----
    gine_init_kernel<<<2048, 256, 0, stream>>>(bx, L.eps, bz, (long long)NH);
    gine_edge_kernel<<<E, H, 0, stream>>>(bx, L.bond, e_src, e_dst, edge_attr, bz);

    // h = relu(bn(z @ l1_w + l1_b));  x = relu(bn(h @ l2_w + l2_b))
    launch_cvt(bz, abf, (long long)NH, stream);
    launch_gemm(abf, Wt.l1, L.l1_b, nullptr, nullptr, bh, N, 2 * H, H, 0, 0, 0, stream);
    bn_stats_kernel<<<2 * H, 256, 0, stream>>>(bh, N, 2 * H, bmean, brstd);
    bn_apply_kernel<<<N, 2 * H, 0, stream>>>(bh, bmean, brstd, L.gbn_g, L.gbn_b, bh, 2 * H);
    launch_cvt(bh, abf, (long long)N2H, stream);
    launch_gemm(abf, Wt.l2, L.l2_b, nullptr, nullptr, bz, N, H, 2 * H, 0, 0, 0, stream);
    bn_stats_kernel<<<H, 256, 0, stream>>>(bz, N, H, bmean, brstd);
    bn_apply_kernel<<<N, H, 0, stream>>>(bz, bmean, brstd, L.bn_g, L.bn_b, bx, H);

    // ---- atoms -> paths + residual conv block ----
    for (int t = 0; t < 2; ++t) {
      const PathP& P = L.pk[t];
      const PathW& PW = Wt.pk[t];
      const int k = t + 3;
      const int S = slots[t];
      const long long SH = (long long)S * H;
      // sc = seg_mean(x[row], col, S)
      launch_zero(bsc, SH, stream);
      launch_zero(bcnt, S, stream);
      scatter_mean_kernel<<<nmap[t], H, 0, stream>>>(bx, arow[t], acol[t], bsc, bcnt);
      seg_div_kernel<<<S, H, 0, stream>>>(bsc, bcnt);
      // xp = xp + relu(sc @ a2p)
      launch_cvt(bsc, abf, SH, stream);
      launch_gemm(abf, PW.a2p, P.a2p_b, bxp[t], nullptr, bxp[t], S, H, H, 0, 0, 2, stream);
      // hb = relu(conv1(xp)) : 3 shifted taps, SAME zero padding within paths
      launch_cvt(bxp[t], abf, SH, stream);
      launch_gemm(abf, PW.c1,          nullptr, nullptr, nullptr, bsc, S, H, H, k, -1, 0, stream);
      launch_gemm(abf, PW.c1 + 1 * HH, nullptr, nullptr, bsc,     bsc, S, H, H, k,  0, 3, stream);
      launch_gemm(abf, PW.c1 + 2 * HH, P.c1_b,  nullptr, bsc,     bsc, S, H, H, k, +1, 4, stream);
      // xp = relu(xp + conv2(hb))
      launch_cvt(bsc, abf, SH, stream);
      launch_gemm(abf, PW.c2,          nullptr, nullptr, nullptr, btmp,   S, H, H, k, -1, 0, stream);
      launch_gemm(abf, PW.c2 + 1 * HH, nullptr, nullptr, btmp,    btmp,   S, H, H, k,  0, 3, stream);
      launch_gemm(abf, PW.c2 + 2 * HH, P.c2_b,  bxp[t],  btmp,    bxp[t], S, H, H, k, +1, 5, stream);
    }

    // ---- paths -> atoms: x += relu(seg_mean(xp[col], row, N) @ p2a) ----
    for (int t = 0; t < 2; ++t) {
      const PathP& P = L.pk[t];
      const PathW& PW = Wt.pk[t];
      launch_zero(bsc, (long long)NH, stream);
      launch_zero(bcnt, N, stream);
      scatter_mean_kernel<<<nmap[t], H, 0, stream>>>(bxp[t], acol[t], arow[t], bsc, bcnt);
      seg_div_kernel<<<N, H, 0, stream>>>(bsc, bcnt);
      launch_cvt(bsc, abf, (long long)NH, stream);
      launch_gemm(abf, PW.p2a, P.p2a_b, bx, nullptr, bx, N, H, H, 0, 0, 2, stream);
    }
  }

  // ---- graph readout + head ----
  launch_zero(bgs, (long long)GRAPHS * H, stream);
  launch_zero(bgc, GRAPHS, stream);
  seg_rows_kernel<<<N, H, 0, stream>>>(bx, batch, bgs, bgc);
  seg_div_kernel<<<GRAPHS, H, 0, stream>>>(bgs, bgc);
  launch_cvt(bgs, abf, (long long)GRAPHS * H, stream);
  launch_gemm(abf, w_atomlin, atom_lin_b, nullptr, nullptr, bxg,
              GRAPHS, H, H, 0, 0, 1, stream);
  launch_cvt(bxg, abf, (long long)GRAPHS * H, stream);
  launch_gemm(abf, w_lin, lin_b, nullptr, nullptr, (float*)d_out,
              GRAPHS, OUT, H, 0, 0, 0, stream);
}
